// FixedSparseLinear_1666447311096
// MI455X (gfx1250) — matrix operand order repaired and checked by reading, not compiled, not run on hardware
//
#include <hip/hip_runtime.h>
#include <hip/hip_bf16.h>
#include <stdint.h>

// ---------------------------------------------------------------------------
// FixedSparseLinear on MI455X (gfx1250):
//   phase 1: zero dense W (64 MB fp32) in workspace
//   phase 2: scatter-add COO values via global_atomic_add_f32
//   phase 3: split W and X into bf16 hi/lo pairs (one-time, off the hot path)
//   phase 4: dense GEMM y = x*W^T + b with v_wmma_f32_16x16x32_bf16, inner
//            loop streams pre-split bf16 tiles global->LDS with
//            GLOBAL_LOAD_ASYNC_TO_LDS_B128 (ASYNCcnt), double buffered.
//   Fallback (small ws): round-1 style GEMM with on-the-fly hi/lo split.
// ---------------------------------------------------------------------------

typedef __attribute__((ext_vector_type(16))) __bf16 v16bf;
typedef __attribute__((ext_vector_type(8)))  __bf16 v8bf;
typedef __attribute__((ext_vector_type(4)))  __bf16 v4bf;
typedef __attribute__((ext_vector_type(8)))  float  v8f;
typedef __attribute__((ext_vector_type(4)))  float  v4f;

#define IN_F   4096
#define OUT_F  4096
#define M_ROWS 8192          // 4 * 2048
#define BK     32            // K-chunk (one bf16 WMMA K)

union Frag16 { v16bf v; v8bf h[2]; };

// ---------------- phase 1: zero W ------------------------------------------
__global__ void fsl_zero_w(float* __restrict__ w, int n4) {
  int i = blockIdx.x * blockDim.x + threadIdx.x;
  int stride = gridDim.x * blockDim.x;
  v4f z = {};
  for (; i < n4; i += stride)
    *(v4f*)(w + (size_t)i * 4) = z;
}

// ---------------- phase 2: COO scatter-add ---------------------------------
__global__ void fsl_scatter(const float* __restrict__ vals,
                            const long long* __restrict__ idx,
                            float* __restrict__ w, int nnz) {
  int i = blockIdx.x * blockDim.x + threadIdx.x;
  if (i < nnz)
    unsafeAtomicAdd(&w[(size_t)idx[i]], vals[i]);   // global_atomic_add_f32
}

// ---------------- phase 3: fp32 -> bf16 hi/lo split -------------------------
__global__ void fsl_split(const float* __restrict__ src,
                          __bf16* __restrict__ hi, __bf16* __restrict__ lo,
                          int n4) {
  int i = blockIdx.x * blockDim.x + threadIdx.x;
  int stride = gridDim.x * blockDim.x;
  for (; i < n4; i += stride) {
    v4f a = *(const v4f*)(src + (size_t)i * 4);
    v4bf h, l;
    for (int e = 0; e < 4; ++e) {
      __bf16 hh = (__bf16)a[e];
      h[e] = hh;
      l[e] = (__bf16)(a[e] - (float)hh);
    }
    *(v4bf*)(hi + (size_t)i * 4) = h;
    *(v4bf*)(lo + (size_t)i * 4) = l;
  }
}

// ---------------- phase 4a: GEMM, async-LDS fast path ----------------------
// Workgroup: 256 threads = 8 wave32, output tile 128(M) x 128(N).
// Wave grid 4(M) x 2(N): each wave computes 32x64 = 2x4 tiles of 16x16.
__global__ __launch_bounds__(256)
void fsl_gemm_async(const __bf16* __restrict__ Xhi, const __bf16* __restrict__ Xlo,
                    const __bf16* __restrict__ Whi, const __bf16* __restrict__ Wlo,
                    const float* __restrict__ bias, float* __restrict__ Y) {
  // [buffer][array: Ahi, Alo, Bhi, Blo][128 rows x 32 k]
  __shared__ alignas(16) __bf16 smem[2][4][128 * BK];

  const int tid   = threadIdx.x;
  const int nbase = blockIdx.x * 128;
  const int mbase = blockIdx.y * 128;
  const int wave  = tid >> 5;
  const int lane  = tid & 31;
  const int wm    = wave >> 1;   // 0..3 -> 32-row strip
  const int wn    = wave & 1;    // 0..1 -> 64-col strip
  const int half  = lane >> 4;
  const int l16   = lane & 15;

  const uint64_t xh = (uint64_t)(uintptr_t)Xhi;
  const uint64_t xl = (uint64_t)(uintptr_t)Xlo;
  const uint64_t wh = (uint64_t)(uintptr_t)Whi;
  const uint64_t wl = (uint64_t)(uintptr_t)Wlo;

  // Issue 8 async b128 global->LDS copies for one K-chunk (this thread's share).
  auto issue = [&](int buf, int kb) {
    uint32_t lds0 = (uint32_t)(uintptr_t)(&smem[buf][0][0]);
    for (int i = 0; i < 2; ++i) {
      int pos = tid + i * 256;          // 0..511 b128 slots per array
      int r   = pos >> 2;               // tile row 0..127
      int c   = pos & 3;                // 16B column
      uint32_t loff = (uint32_t)(r * 64 + c * 16);
      uint32_t gx = (uint32_t)(((mbase + r) * IN_F + kb + c * 8) * 2);
      uint32_t gw = (uint32_t)(((nbase + r) * IN_F + kb + c * 8) * 2);
      asm volatile("global_load_async_to_lds_b128 %0, %1, %2"
                   :: "v"(lds0 + 0 * 8192 + loff), "v"(gx), "s"(xh) : "memory");
      asm volatile("global_load_async_to_lds_b128 %0, %1, %2"
                   :: "v"(lds0 + 1 * 8192 + loff), "v"(gx), "s"(xl) : "memory");
      asm volatile("global_load_async_to_lds_b128 %0, %1, %2"
                   :: "v"(lds0 + 2 * 8192 + loff), "v"(gw), "s"(wh) : "memory");
      asm volatile("global_load_async_to_lds_b128 %0, %1, %2"
                   :: "v"(lds0 + 3 * 8192 + loff), "v"(gw), "s"(wl) : "memory");
    }
  };

  v8f acc[2][4];
  for (int t = 0; t < 2; ++t)
    for (int g = 0; g < 4; ++g)
      acc[t][g] = v8f{};

  issue(0, 0);
  asm volatile("s_wait_asynccnt 0" ::: "memory");
  __syncthreads();

  int cur = 0;
  for (int kb = 0; kb < IN_F; kb += BK) {
    if (kb + BK < IN_F) issue(cur ^ 1, kb + BK);   // prefetch next chunk (DMA)

    const __bf16* Ah = &smem[cur][0][0];
    const __bf16* Al = &smem[cur][1][0];
    const __bf16* Bh = &smem[cur][2][0];
    const __bf16* Bl = &smem[cur][3][0];

    // A fragments (16-bit A 16x32): lanes0-15 row=l16 K=0..7 & 16..23;
    //                               lanes16-31 row=l16 K=8..15 & 24..31
    Frag16 afh[2], afl[2];
    for (int t = 0; t < 2; ++t) {
      int row = (wm * 32 + t * 16 + l16) * BK;
      int k0  = half * 8;
      afh[t].h[0] = *(const v8bf*)(Ah + row + k0);
      afh[t].h[1] = *(const v8bf*)(Ah + row + k0 + 16);
      afl[t].h[0] = *(const v8bf*)(Al + row + k0);
      afl[t].h[1] = *(const v8bf*)(Al + row + k0 + 16);
    }
    // B fragments (16-bit B 32x16): lanes0-15 N=l16 K=0..7 & 16..23; lanes16-31 K=8..15 & 24..31
    // B[k][n] = W[n][k] -> native [n][k] tile is k-contiguous.
    Frag16 bfh[4], bfl[4];
    for (int g = 0; g < 4; ++g) {
      int col = (wn * 64 + g * 16 + l16) * BK;
      int k0  = half * 8;
      bfh[g].h[0] = *(const v8bf*)(Bh + col + k0);
      bfh[g].h[1] = *(const v8bf*)(Bh + col + k0 + 16);
      bfl[g].h[0] = *(const v8bf*)(Bl + col + k0);
      bfl[g].h[1] = *(const v8bf*)(Bl + col + k0 + 16);
    }

    for (int t = 0; t < 2; ++t)
      for (int g = 0; g < 4; ++g) {
        acc[t][g] = __builtin_amdgcn_wmma_f32_16x16x32_bf16(
            false, afh[t].v, false, bfh[g].v, (short)0, acc[t][g], false, false);
        acc[t][g] = __builtin_amdgcn_wmma_f32_16x16x32_bf16(
            false, afh[t].v, false, bfl[g].v, (short)0, acc[t][g], false, false);
        acc[t][g] = __builtin_amdgcn_wmma_f32_16x16x32_bf16(
            false, afl[t].v, false, bfh[g].v, (short)0, acc[t][g], false, false);
      }

    asm volatile("s_wait_asynccnt 0" ::: "memory");
    __syncthreads();
    cur ^= 1;
  }

  // Epilogue: C 16x16 f32 layout: VGPR r -> M = half*8 + r, N = l16
  for (int t = 0; t < 2; ++t)
    for (int g = 0; g < 4; ++g) {
      int col  = nbase + wn * 64 + g * 16 + l16;
      float bv = bias[col];
      int rowb = mbase + wm * 32 + t * 16 + half * 8;
      for (int r = 0; r < 8; ++r)
        Y[(size_t)(rowb + r) * OUT_F + col] = acc[t][g][r] + bv;
    }
}

// ---------------- phase 4b: GEMM fallback (on-the-fly split) ---------------
__global__ __launch_bounds__(256)
void fsl_gemm(const float* __restrict__ X, const float* __restrict__ W,
              const float* __restrict__ bias, float* __restrict__ Y) {
  __shared__ alignas(16) __bf16 Ahi[128 * BK];
  __shared__ alignas(16) __bf16 Alo[128 * BK];
  __shared__ alignas(16) __bf16 Bhi[128 * BK];
  __shared__ alignas(16) __bf16 Blo[128 * BK];

  const int tid   = threadIdx.x;
  const int nbase = blockIdx.x * 128;
  const int mbase = blockIdx.y * 128;
  const int wave  = tid >> 5;
  const int lane  = tid & 31;
  const int wm    = wave >> 1;
  const int wn    = wave & 1;
  const int half  = lane >> 4;
  const int l16   = lane & 15;

  v8f acc[2][4];
  for (int t = 0; t < 2; ++t)
    for (int g = 0; g < 4; ++g)
      acc[t][g] = v8f{};

  for (int kb = 0; kb < IN_F; kb += BK) {
    for (int i = 0; i < 4; ++i) {
      int pos = tid + i * 256;
      int r   = pos >> 3;
      int c4  = pos & 7;
      v4f a = *(const v4f*)(X + (size_t)(mbase + r) * IN_F + kb + c4 * 4);
      v4f b = *(const v4f*)(W + (size_t)(nbase + r) * IN_F + kb + c4 * 4);
      v4bf ah, al, bh, bl;
      for (int e = 0; e < 4; ++e) {
        __bf16 h0 = (__bf16)a[e];
        ah[e] = h0; al[e] = (__bf16)(a[e] - (float)h0);
        __bf16 h1 = (__bf16)b[e];
        bh[e] = h1; bl[e] = (__bf16)(b[e] - (float)h1);
      }
      *(v4bf*)(Ahi + r * BK + c4 * 4) = ah;
      *(v4bf*)(Alo + r * BK + c4 * 4) = al;
      *(v4bf*)(Bhi + r * BK + c4 * 4) = bh;
      *(v4bf*)(Blo + r * BK + c4 * 4) = bl;
    }
    __syncthreads();

    Frag16 afh[2], afl[2];
    for (int t = 0; t < 2; ++t) {
      int row = (wm * 32 + t * 16 + l16) * BK;
      int k0  = half * 8;
      afh[t].h[0] = *(const v8bf*)(Ahi + row + k0);
      afh[t].h[1] = *(const v8bf*)(Ahi + row + k0 + 16);
      afl[t].h[0] = *(const v8bf*)(Alo + row + k0);
      afl[t].h[1] = *(const v8bf*)(Alo + row + k0 + 16);
    }
    Frag16 bfh[4], bfl[4];
    for (int g = 0; g < 4; ++g) {
      int col = (wn * 64 + g * 16 + l16) * BK;
      int k0  = half * 8;
      bfh[g].h[0] = *(const v8bf*)(Bhi + col + k0);
      bfh[g].h[1] = *(const v8bf*)(Bhi + col + k0 + 16);
      bfl[g].h[0] = *(const v8bf*)(Blo + col + k0);
      bfl[g].h[1] = *(const v8bf*)(Blo + col + k0 + 16);
    }
    for (int t = 0; t < 2; ++t)
      for (int g = 0; g < 4; ++g) {
        acc[t][g] = __builtin_amdgcn_wmma_f32_16x16x32_bf16(
            false, afh[t].v, false, bfh[g].v, (short)0, acc[t][g], false, false);
        acc[t][g] = __builtin_amdgcn_wmma_f32_16x16x32_bf16(
            false, afh[t].v, false, bfl[g].v, (short)0, acc[t][g], false, false);
        acc[t][g] = __builtin_amdgcn_wmma_f32_16x16x32_bf16(
            false, afl[t].v, false, bfh[g].v, (short)0, acc[t][g], false, false);
      }
    __syncthreads();
  }

  for (int t = 0; t < 2; ++t)
    for (int g = 0; g < 4; ++g) {
      int col  = nbase + wn * 64 + g * 16 + l16;
      float bv = bias[col];
      int rowb = mbase + wm * 32 + t * 16 + half * 8;
      for (int r = 0; r < 8; ++r)
        Y[(size_t)(rowb + r) * OUT_F + col] = acc[t][g][r] + bv;
    }
}

// ---------------------------------------------------------------------------
extern "C" void kernel_launch(void* const* d_in, const int* in_sizes, int n_in,
                              void* d_out, int out_size, void* d_ws, size_t ws_size,
                              hipStream_t stream) {
  const float*     x    = (const float*)d_in[0];
  const float*     vals = (const float*)d_in[1];
  const float*     bias = (const float*)d_in[2];
  const long long* idx  = (const long long*)d_in[3];
  float* y = (float*)d_out;
  const int nnz = in_sizes[1];

  char* ws = (char*)d_ws;
  float* wf32 = (float*)ws;                               // 64 MB
  const size_t W_ELEMS = (size_t)OUT_F * IN_F;            // 16.8M
  const size_t X_ELEMS = (size_t)M_ROWS * IN_F;           // 33.6M

  // phase 1: clear W (workspace is poisoned; must be zeroed every call)
  fsl_zero_w<<<2048, 256, 0, stream>>>(wf32, (int)(W_ELEMS / 4));
  // phase 2: scatter-add
  fsl_scatter<<<(nnz + 255) / 256, 256, 0, stream>>>(vals, idx, wf32, nnz);

  const size_t need = W_ELEMS * 4           // wf32
                    + W_ELEMS * 2 * 2       // whi + wlo
                    + X_ELEMS * 2 * 2;      // xhi + xlo   (256 MB total)
  if (ws_size >= need) {
    __bf16* whi = (__bf16*)(ws + W_ELEMS * 4);
    __bf16* wlo = whi + W_ELEMS;
    __bf16* xhi = (__bf16*)(ws + W_ELEMS * 4 + W_ELEMS * 4);
    __bf16* xlo = xhi + X_ELEMS;
    // phase 3: one-time hi/lo splits
    fsl_split<<<4096, 256, 0, stream>>>(wf32, whi, wlo, (int)(W_ELEMS / 4));
    fsl_split<<<8192, 256, 0, stream>>>(x,    xhi, xlo, (int)(X_ELEMS / 4));
    // phase 4: async-LDS GEMM  grid = (N/128, M/128)
    dim3 grid(OUT_F / 128, M_ROWS / 128);
    fsl_gemm_async<<<grid, 256, 0, stream>>>(xhi, xlo, whi, wlo, bias, y);
  } else {
    dim3 grid(OUT_F / 128, M_ROWS / 128);
    fsl_gemm<<<grid, 256, 0, stream>>>(x, wf32, bias, y);
  }
}
